// DiTAttention_53446573031484
// MI455X (gfx1250) — matrix operand order repaired and checked by reading, not compiled, not run on hardware
//
#include <hip/hip_runtime.h>

#define DEVI __device__ __forceinline__

typedef __bf16 bf16_t;
typedef bf16_t v16bf __attribute__((ext_vector_type(16)));
typedef bf16_t v8bf  __attribute__((ext_vector_type(8)));
typedef float  v8f   __attribute__((ext_vector_type(8)));

namespace {
constexpr int BATCH = 2;
constexpr int SEQ   = 2048;
constexpr int NDIM  = 1024;
constexpr int HEADS = 16;
constexpr int HD    = 64;
constexpr int MTOT  = BATCH * SEQ;      // 4096
constexpr float SM_SCALE = 0.125f;      // 1/sqrt(64)
}

// ---------- helpers -------------------------------------------------------

DEVI unsigned short f2bf(float f) {                 // fp32 -> bf16 RNE
    unsigned int u = __float_as_uint(f);
    u += 0x7FFFu + ((u >> 16) & 1u);
    return (unsigned short)(u >> 16);
}

DEVI v8bf ld8(const unsigned short* p) { return *(const v8bf*)p; }

DEVI v16bf join16(v8bf lo, v8bf hi) {
    return __builtin_shufflevector(lo, hi, 0,1,2,3,4,5,6,7,8,9,10,11,12,13,14,15);
}

DEVI v8f vzero() {
    v8f z;
#pragma unroll
    for (int r = 0; r < 8; ++r) z[r] = 0.0f;
    return z;
}

// A-matrix fragment (16x32 bf16) from row-major [rows, ld] bf16 buffer.
// ISA layout: lanes 0-15 -> M=lane, K {0..7,16..23}; lanes 16-31 -> M=lane-16,
// K {8..15,24..31}. Two 16B loads per lane.
DEVI v16bf loadA(const unsigned short* base, int row0, int ld, int k0, int lane) {
    const int half = lane >> 4;
    const int m    = row0 + (lane & 15);
    const unsigned short* p = base + (size_t)m * ld + k0 + half * 8;
    return join16(ld8(p), ld8(p + 16));
}

// Same layout but from a per-wave LDS tile with row stride 32 halves.
DEVI v16bf loadA_lds(const unsigned short* tile, int lane) {
    const int half = lane >> 4;
    const unsigned short* p = tile + (lane & 15) * 32 + half * 8;
    return join16(ld8(p), ld8(p + 16));
}

// B-matrix fragment (32x16 bf16) where B[k][n] = src[(n0+n)*ld + k0 + k].
// ISA layout: lanes 0-15 -> N=lane, K {0..7,16..23}; lanes 16-31 -> N=lane-16,
// K {8..15,24..31}. Two 16B loads per lane.
DEVI v16bf loadB(const unsigned short* base, int n0, int ld, int k0, int lane) {
    const int half = lane >> 4;
    const int n    = n0 + (lane & 15);
    const unsigned short* p = base + (size_t)n * ld + k0 + half * 8;
    return join16(ld8(p), ld8(p + 16));
}

DEVI v8f wmma_bf16(v16bf a, v16bf b, v8f c) {
    return __builtin_amdgcn_wmma_f32_16x16x32_bf16(
        /*neg_a=*/false, a, /*neg_b=*/false, b,
        /*c_mod=*/(short)0, c, /*reuse_a=*/false, /*reuse_b=*/false);
}

// Async DMA: one 16B chunk global -> LDS, tracked by ASYNCcnt (no VGPR data).
DEVI void async_copy16(const unsigned short* gptr, unsigned short* lptr) {
    const unsigned lds_off = (unsigned)(unsigned long long)(void*)lptr;
    asm volatile("global_load_async_to_lds_b128 %0, %1, off"
                 :: "v"(lds_off), "v"(gptr) : "memory");
}

DEVI void wait_async0() {
    asm volatile("s_wait_asynccnt 0" ::: "memory");
}

// ---------- fp32 -> bf16 convert ------------------------------------------

__global__ void cvt_f32_bf16(const float* __restrict__ src,
                             unsigned short* __restrict__ dst, int n) {
    int i = blockIdx.x * blockDim.x + threadIdx.x;
    if (i < n) dst[i] = f2bf(src[i]);
}

// ---------- WMMA GEMM: out = A[M,K] * W[N,K]^T + bias ---------------------
// MODE 0: Q  -> bf16 [B,H,S,D], RoPE on head 0
// MODE 1: K  -> bf16 [B,H,S,D], RoPE on head 0
// MODE 2: V  -> bf16 [B,H,D,S] (transposed for PV B-operand)
// MODE 3: O  -> fp32 [M, N]
template <int MODE>
__global__ __launch_bounds__(256) void gemm_bf16(
    const unsigned short* __restrict__ A,    // bf16 [MTOT, NDIM]
    const unsigned short* __restrict__ W,    // bf16 [NDIM, NDIM] (row-major [N,K])
    const float* __restrict__ bias,          // [NDIM]
    const float* __restrict__ cosb,          // [SEQ, HD]
    const float* __restrict__ sinb,          // [SEQ, HD]
    unsigned short* __restrict__ obf,        // bf16 out (modes 0..2)
    float* __restrict__ ofp)                 // fp32 out (mode 3)
{
    constexpr int K = NDIM;
    const int lane = threadIdx.x & 31;
    const int wv   = threadIdx.x >> 5;                  // 8 waves
    const int m0   = blockIdx.y * 128 + (wv >> 2) * 64; // 2 waves along M
    const int n0   = blockIdx.x * 128 + (wv & 3) * 32;  // 4 waves along N

    v8f acc[4][2];
#pragma unroll
    for (int i = 0; i < 4; ++i)
#pragma unroll
        for (int j = 0; j < 2; ++j) acc[i][j] = vzero();

    for (int k0 = 0; k0 < K; k0 += 32) {
        v16bf a[4], b[2];
#pragma unroll
        for (int i = 0; i < 4; ++i) a[i] = loadA(A, m0 + 16 * i, K, k0, lane);
#pragma unroll
        for (int j = 0; j < 2; ++j) b[j] = loadB(W, n0 + 16 * j, K, k0, lane);
#pragma unroll
        for (int i = 0; i < 4; ++i)
#pragma unroll
            for (int j = 0; j < 2; ++j) acc[i][j] = wmma_bf16(a[i], b[j], acc[i][j]);
    }

    // Epilogue.  C layout: VGPR r, lanes 0-15 -> (M=r, N=lane),
    // lanes 16-31 -> (M=r+8, N=lane-16).
    const int colOff = lane & 15;
    const int rowOff = (lane >> 4) * 8;
#pragma unroll
    for (int i = 0; i < 4; ++i) {
#pragma unroll
        for (int j = 0; j < 2; ++j) {
            const int nb = n0 + 16 * j;
            const int n  = nb + colOff;
#pragma unroll
            for (int r = 0; r < 8; ++r) {
                const int m = m0 + 16 * i + rowOff + r;
                const int s = m & (SEQ - 1);
                const int bi = m >> 11;                   // m / SEQ
                float val = acc[i][j][r] + bias[n];
                if (MODE == 0 || MODE == 1) {
                    // interleaved-pair RoPE, head 0 only (nb<HD is fragment-uniform)
                    float partner = __shfl_xor(val, 1);
                    if (nb < HD) {
                        float c  = cosb[s * HD + n];
                        float sn = sinb[s * HD + n];
                        val = (n & 1) ? (val * c + partner * sn)
                                      : (val * c - partner * sn);
                    }
                }
                if (MODE == 3) {
                    ofp[(size_t)m * NDIM + n] = val;
                } else {
                    const int h = n >> 6, d = n & 63;
                    if (MODE == 2) {  // V^T : [B,H,D,S]
                        obf[(((size_t)bi * HEADS + h) * HD + d) * SEQ + s] = f2bf(val);
                    } else {          // Q/K : [B,H,S,D]
                        obf[(((size_t)bi * HEADS + h) * SEQ + s) * HD + d] = f2bf(val);
                    }
                }
            }
        }
    }
}

// ---------- flash attention -----------------------------------------------
// K/V tiles staged once per workgroup via async global->LDS DMA (double
// buffered), consumed by all 8 waves from LDS; bf16 WMMA, fp32 online softmax.

__global__ __launch_bounds__(256) void attn_kernel(
    const unsigned short* __restrict__ Q,    // bf16 [B,H,S,D]
    const unsigned short* __restrict__ Kb,   // bf16 [B,H,S,D]
    const unsigned short* __restrict__ Vt,   // bf16 [B,H,D,S]
    unsigned short* __restrict__ ctx)        // bf16 [B,S,H*D]
{
    __shared__ __align__(16) unsigned short kt[2][32 * HD];   // 2 x 4KB key tiles
    __shared__ __align__(16) unsigned short vtl[2][HD * 32];  // 2 x 4KB V^T tiles
    __shared__ __align__(16) unsigned short pbuf[8 * 16 * 32];// per-wave P tiles

    const int tid  = threadIdx.x;
    const int lane = tid & 31;
    const int wv   = tid >> 5;
    const int blocksPerBH = SEQ / 128;        // 16
    const int bh = blockIdx.x / blocksPerBH;
    const int qt = blockIdx.x % blocksPerBH;
    const int b  = bh / HEADS;
    const int h  = bh % HEADS;

    const unsigned short* Qp = Q  + ((size_t)(b * HEADS + h) * SEQ) * HD;
    const unsigned short* Kp = Kb + ((size_t)(b * HEADS + h) * SEQ) * HD;
    const unsigned short* Vp = Vt + ((size_t)(b * HEADS + h) * HD) * SEQ;

    const int q0 = qt * 128 + wv * 16;        // 16 query rows per wave
    const int rowOff = (lane >> 4) * 8;
    const int colOff = lane & 15;

    // Stage one 32-key tile (K: contiguous 4KB; V^T: 64 rows x 64B) — one
    // 16B async chunk per thread per tile.
    auto stageK = [&](int kb, int buf) {
        async_copy16(Kp + (size_t)kb * HD + tid * 8, &kt[buf][tid * 8]);
    };
    auto stageV = [&](int kb, int buf) {
        const int d = tid >> 2, part = tid & 3;
        async_copy16(Vp + (size_t)d * SEQ + kb + part * 8,
                     &vtl[buf][d * 32 + part * 8]);
    };

    // Q fragments for head-dim halves, resident all kernel
    const v16bf qa0 = loadA(Qp, q0, HD, 0,  lane);
    const v16bf qa1 = loadA(Qp, q0, HD, 32, lane);

    v8f o[4];
#pragma unroll
    for (int t = 0; t < 4; ++t) o[t] = vzero();
    float mrow[8], lrow[8];
#pragma unroll
    for (int r = 0; r < 8; ++r) { mrow[r] = -1e30f; lrow[r] = 0.0f; }

    unsigned short* pw = pbuf + wv * 16 * 32;

    stageK(0, 0);
    stageV(0, 0);
    wait_async0();
    __syncthreads();

    for (int kb = 0; kb < SEQ; kb += 32) {
        const int cur = (kb >> 5) & 1;
        const int nxt = cur ^ 1;
        if (kb + 32 < SEQ) {          // uniform branch: prefetch next tile
            stageK(kb + 32, nxt);
            stageV(kb + 32, nxt);
        }
        const unsigned short* kl = kt[cur];
        const unsigned short* vl = vtl[cur];

        // S tile (16 rows x 32 keys) = Q * K^T  — 4 WMMAs fed from LDS
        v8f s0 = vzero(), s1 = vzero();
        {
            v16bf bk;
            bk = loadB(kl, 0,  HD, 0,  lane); s0 = wmma_bf16(qa0, bk, s0);
            bk = loadB(kl, 0,  HD, 32, lane); s0 = wmma_bf16(qa1, bk, s0);
            bk = loadB(kl, 16, HD, 0,  lane); s1 = wmma_bf16(qa0, bk, s1);
            bk = loadB(kl, 16, HD, 32, lane); s1 = wmma_bf16(qa1, bk, s1);
        }

        // online softmax: row = rowOff + r lives on 16 lanes of this half
        float mnew[8], alpha[8];
#pragma unroll
        for (int r = 0; r < 8; ++r) {
            float a = s0[r] * SM_SCALE, bb = s1[r] * SM_SCALE;
            s0[r] = a; s1[r] = bb;
            float mx = fmaxf(a, bb);
            mx = fmaxf(mx, __shfl_xor(mx, 1));
            mx = fmaxf(mx, __shfl_xor(mx, 2));
            mx = fmaxf(mx, __shfl_xor(mx, 4));
            mx = fmaxf(mx, __shfl_xor(mx, 8));
            mnew[r]  = fmaxf(mrow[r], mx);
            alpha[r] = __expf(mrow[r] - mnew[r]);
            mrow[r]  = mnew[r];
        }

        // exp, row-sums, rescale O, spill P (bf16) to LDS in row-major 16x32
#pragma unroll
        for (int r = 0; r < 8; ++r) {
            float p0 = __expf(s0[r] - mnew[r]);
            float p1 = __expf(s1[r] - mnew[r]);
            const int row = rowOff + r;
            pw[row * 32 + colOff]      = f2bf(p0);
            pw[row * 32 + colOff + 16] = f2bf(p1);
            float rs = p0 + p1;
            rs += __shfl_xor(rs, 1);
            rs += __shfl_xor(rs, 2);
            rs += __shfl_xor(rs, 4);
            rs += __shfl_xor(rs, 8);
            lrow[r] = lrow[r] * alpha[r] + rs;
            o[0][r] *= alpha[r]; o[1][r] *= alpha[r];
            o[2][r] *= alpha[r]; o[3][r] *= alpha[r];
        }

        // P (16x32) as A-fragment from LDS; in-wave DS ordering suffices
        const v16bf pa = loadA_lds(pw, lane);

        // O += P * V  — 4 WMMAs, B from V^T tile in LDS (row stride 32)
#pragma unroll
        for (int t = 0; t < 4; ++t) {
            v16bf bv = loadB(vl, 16 * t, 32, 0, lane);
            o[t] = wmma_bf16(pa, bv, o[t]);
        }

        // drain this wave's DMA into `nxt`, then publish tile to all waves;
        // also guarantees everyone is done reading `cur` before it is refilled.
        wait_async0();
        __syncthreads();
    }

    // finalize: divide by l, scatter to ctx[B,S,H*D]
#pragma unroll
    for (int t = 0; t < 4; ++t) {
#pragma unroll
        for (int r = 0; r < 8; ++r) {
            const int srow = q0 + rowOff + r;
            const int d    = 16 * t + colOff;
            const float val = o[t][r] / lrow[r];
            ctx[((size_t)b * SEQ + srow) * NDIM + h * HD + d] = f2bf(val);
        }
    }
}

// ---------- launch --------------------------------------------------------

extern "C" void kernel_launch(void* const* d_in, const int* in_sizes, int n_in,
                              void* d_out, int out_size, void* d_ws, size_t ws_size,
                              hipStream_t stream) {
    const float* x    = (const float*)d_in[0];
    const float* cosb = (const float*)d_in[1];   // [1,S,1,64] -> flat [S,64]
    const float* sinb = (const float*)d_in[2];
    const float* Wq = (const float*)d_in[3]; const float* bq = (const float*)d_in[4];
    const float* Wk = (const float*)d_in[5]; const float* bk = (const float*)d_in[6];
    const float* Wv = (const float*)d_in[7]; const float* bv = (const float*)d_in[8];
    const float* Wo = (const float*)d_in[9]; const float* bo = (const float*)d_in[10];
    // d_in[11] is the mask: all-true in this model -> additive term is 0, omitted.

    char* p = (char*)d_ws;
    unsigned short* xbf = (unsigned short*)p; p += (size_t)MTOT * NDIM * 2;
    unsigned short* wqb = (unsigned short*)p; p += (size_t)NDIM * NDIM * 2;
    unsigned short* wkb = (unsigned short*)p; p += (size_t)NDIM * NDIM * 2;
    unsigned short* wvb = (unsigned short*)p; p += (size_t)NDIM * NDIM * 2;
    unsigned short* wob = (unsigned short*)p; p += (size_t)NDIM * NDIM * 2;
    unsigned short* Qb  = (unsigned short*)p; p += (size_t)MTOT * NDIM * 2;
    unsigned short* Kbf = (unsigned short*)p; p += (size_t)MTOT * NDIM * 2;
    unsigned short* Vtb = (unsigned short*)p; p += (size_t)MTOT * NDIM * 2;
    unsigned short* ctx = xbf;   // x-bf16 is dead after the V GEMM; reuse for ctx

    const int nX = MTOT * NDIM;       // 4,194,304
    const int nW = NDIM * NDIM;       // 1,048,576
    cvt_f32_bf16<<<nX / 256, 256, 0, stream>>>(x,  xbf, nX);
    cvt_f32_bf16<<<nW / 256, 256, 0, stream>>>(Wq, wqb, nW);
    cvt_f32_bf16<<<nW / 256, 256, 0, stream>>>(Wk, wkb, nW);
    cvt_f32_bf16<<<nW / 256, 256, 0, stream>>>(Wv, wvb, nW);
    cvt_f32_bf16<<<nW / 256, 256, 0, stream>>>(Wo, wob, nW);

    dim3 gg(NDIM / 128, MTOT / 128);  // (8, 32)
    gemm_bf16<0><<<gg, 256, 0, stream>>>(xbf, wqb, bq, cosb, sinb, Qb,  nullptr);
    gemm_bf16<1><<<gg, 256, 0, stream>>>(xbf, wkb, bk, cosb, sinb, Kbf, nullptr);
    gemm_bf16<2><<<gg, 256, 0, stream>>>(xbf, wvb, bv, cosb, sinb, Vtb, nullptr);

    attn_kernel<<<BATCH * HEADS * (SEQ / 128), 256, 0, stream>>>(Qb, Kbf, Vtb, ctx);

    gemm_bf16<3><<<gg, 256, 0, stream>>>(ctx, wob, bo, cosb, sinb, nullptr,
                                         (float*)d_out);
}
